// LocalAggregationLoss_33603824124604
// MI455X (gfx1250) — matrix operand order repaired and checked by reading, not compiled, not run on hardware
//
#include <hip/hip_runtime.h>
#include <math.h>

typedef __attribute__((ext_vector_type(16))) _Float16 v16h;
typedef __attribute__((ext_vector_type(8)))  float    v8f;
typedef __attribute__((ext_vector_type(4)))  float    v4f;

#define B_ROWS 512
#define N_COLS 100000
#define D_DIM  128
#define INV_T  (1.0f / 0.07f)
#define M_TILES (B_ROWS / 16)   // 32
#define N_TILES (N_COLS / 16)   // 6250

// ws layout (bytes):
//   [0,       2048)  : d1[512] floats
//   [2048,    4096)  : d2[512] floats
//   [4096,    6144)  : inv_norm[512] floats
//   [8192, +25.6MB)  : bankHi  (N*D f16)   -- fast path only
//   [... , +25.6MB)  : bankLo  (N*D f16)   -- fast path only
#define WS_PLANE_OFF   8192
#define WS_FAST_BYTES  (WS_PLANE_OFF + (size_t)2 * N_COLS * D_DIM * 2)

// ---------------------------------------------------------------------------
// Kernel 1: zero d1/d2 accumulators, compute per-row inverse L2 norms of codes
// ---------------------------------------------------------------------------
__global__ void laggr_prep_kernel(const float* __restrict__ codes,
                                  float* __restrict__ ws) {
    int i = blockIdx.x * blockDim.x + threadIdx.x;
    if (i < 2 * B_ROWS) ws[i] = 0.0f;
    if (i < B_ROWS) {
        const float* row = codes + (size_t)i * D_DIM;
        float s = 0.0f;
        for (int k = 0; k < D_DIM; ++k) { float x = row[k]; s += x * x; }
        ws[2 * B_ROWS + i] = 1.0f / sqrtf(s);
    }
}

// ---------------------------------------------------------------------------
// Kernel 2 (fast path): split bank fp32 -> f16 hi/lo planes, once per element.
// ---------------------------------------------------------------------------
__global__ void __launch_bounds__(256)
laggr_split_kernel(const float* __restrict__ bank,
                   _Float16* __restrict__ hi,
                   _Float16* __restrict__ lo) {
    const size_t total4 = (size_t)N_COLS * D_DIM / 4;
    size_t i = (size_t)blockIdx.x * blockDim.x + threadIdx.x;
    const size_t stride = (size_t)gridDim.x * blockDim.x;
    for (; i < total4; i += stride) {
        v4f x = ((const v4f*)bank)[i];
#pragma unroll
        for (int j = 0; j < 4; ++j) {
            _Float16 h = (_Float16)x[j];
            hi[i * 4 + j] = h;
            lo[i * 4 + j] = (_Float16)(x[j] - (float)h);
        }
    }
}

// ---------------------------------------------------------------------------
// Shared epilogue helpers
// ---------------------------------------------------------------------------
__device__ __forceinline__ void laggr_reduce_commit(float* d1, float* d2,
                                                    const float* s1, const float* s2,
                                                    int m_base, int lrow, int hgrp) {
#pragma unroll
    for (int r = 0; r < 8; ++r) {
        float x1 = s1[r], x2 = s2[r];
#pragma unroll
        for (int off = 8; off > 0; off >>= 1) {
            x1 += __shfl_xor(x1, off, 16);
            x2 += __shfl_xor(x2, off, 16);
        }
        if (lrow == 0) {
            const int mm = m_base + r + hgrp * 8;
            atomicAdd(&d1[mm], x1);
            atomicAdd(&d2[mm], x2);
        }
    }
}

// Load+normalize+split one wave's A tile (16 rows x 128 K) into registers.
// A operand, contiguous k order (same as B): lane holds row m=l%16; v16h elems 0..15 = K[kb..kb+15],
// kb = kstep*32 + (l/16)*16.
__device__ __forceinline__ void laggr_load_a(const float* codes, const float* invn,
                                             int m_base, int lrow, int hgrp,
                                             v16h a_hi[4], v16h a_lo[4]) {
    const float* arow = codes + (size_t)(m_base + lrow) * D_DIM;
    const float ascale = invn[m_base + lrow];
#pragma unroll
    for (int ks = 0; ks < 4; ++ks) {
        const int kb = ks * 32 + hgrp * 16;
#pragma unroll
        for (int j = 0; j < 8; ++j) {
            float x0 = arow[kb + j]      * ascale;
            float x1 = arow[kb + 8 + j]  * ascale;
            _Float16 h0 = (_Float16)x0;
            _Float16 h1 = (_Float16)x1;
            a_hi[ks][j]     = h0;
            a_hi[ks][8 + j] = h1;
            a_lo[ks][j]     = (_Float16)(x0 - (float)h0);
            a_lo[ks][8 + j] = (_Float16)(x1 - (float)h1);
        }
    }
}

// ---------------------------------------------------------------------------
// Kernel 3a (fast path): GEMM reading preconverted f16 hi/lo planes.
// Inner loop is pure vmem + v_wmma; only exp/mask epilogue in VALU.
// ---------------------------------------------------------------------------
__global__ void __launch_bounds__(128)
laggr_gemm_fast_kernel(const float* __restrict__ codes,
                       const _Float16* __restrict__ bankHi,
                       const _Float16* __restrict__ bankLo,
                       const unsigned char* __restrict__ bgm,
                       const unsigned char* __restrict__ inm,
                       float* __restrict__ ws) {
    float* d1 = ws;
    float* d2 = ws + B_ROWS;
    const float* invn = ws + 2 * B_ROWS;

    const int lane = threadIdx.x & 31;
    const int lrow = lane & 15;
    const int hgrp = lane >> 4;

    const int wave      = blockIdx.x * (blockDim.x >> 5) + (threadIdx.x >> 5);
    const int tot_waves = gridDim.x * (blockDim.x >> 5);
    const int m_tile    = wave & (M_TILES - 1);
    const int walker    = wave / M_TILES;
    const int n_walkers = tot_waves / M_TILES;
    const int m_base    = m_tile * 16;

    v16h a_hi[4], a_lo[4];
    laggr_load_a(codes, invn, m_base, lrow, hgrp, a_hi, a_lo);

    float s1[8], s2[8];
#pragma unroll
    for (int r = 0; r < 8; ++r) { s1[r] = 0.0f; s2[r] = 0.0f; }

    for (int nt = walker; nt < N_TILES; nt += n_walkers) {
        const int n0 = nt * 16;
        // B operand, contiguous k order (same as A): lane holds column n = n0 + l%16 (= bank row),
        // v16h elems 0..15 = K[kb..kb+15], kb = kstep*32 + (l/16)*16.
        const size_t rbase = (size_t)(n0 + lrow) * D_DIM + hgrp * 16;

        v8f acc = {};
#pragma unroll
        for (int ks = 0; ks < 4; ++ks) {
            v16h b_hi = *(const v16h*)(bankHi + rbase + ks * 32);
            v16h b_lo = *(const v16h*)(bankLo + rbase + ks * 32);
            acc = __builtin_amdgcn_wmma_f32_16x16x32_f16(
                false, a_hi[ks], false, b_hi, (short)0, acc, false, false);
            acc = __builtin_amdgcn_wmma_f32_16x16x32_f16(
                false, a_hi[ks], false, b_lo, (short)0, acc, false, false);
            acc = __builtin_amdgcn_wmma_f32_16x16x32_f16(
                false, a_lo[ks], false, b_hi, (short)0, acc, false, false);
        }

        const int n = n0 + lrow;
#pragma unroll
        for (int r = 0; r < 8; ++r) {
            const int mm = m_base + r + hgrp * 8;
            const size_t idx = (size_t)mm * N_COLS + n;
            const float e = __expf(acc[r] * INV_T);
            s1[r] += e * (float)bgm[idx];
            s2[r] += e * (float)inm[idx];
        }
    }

    laggr_reduce_commit(d1, d2, s1, s2, m_base, lrow, hgrp);
}

// ---------------------------------------------------------------------------
// Kernel 3b (fallback, small ws): on-the-fly bank split (previous round).
// ---------------------------------------------------------------------------
__global__ void __launch_bounds__(128)
laggr_gemm_fallback_kernel(const float* __restrict__ codes,
                           const float* __restrict__ bank,
                           const unsigned char* __restrict__ bgm,
                           const unsigned char* __restrict__ inm,
                           float* __restrict__ ws) {
    float* d1 = ws;
    float* d2 = ws + B_ROWS;
    const float* invn = ws + 2 * B_ROWS;

    const int lane = threadIdx.x & 31;
    const int lrow = lane & 15;
    const int hgrp = lane >> 4;

    const int wave      = blockIdx.x * (blockDim.x >> 5) + (threadIdx.x >> 5);
    const int tot_waves = gridDim.x * (blockDim.x >> 5);
    const int m_tile    = wave & (M_TILES - 1);
    const int walker    = wave / M_TILES;
    const int n_walkers = tot_waves / M_TILES;
    const int m_base    = m_tile * 16;

    v16h a_hi[4], a_lo[4];
    laggr_load_a(codes, invn, m_base, lrow, hgrp, a_hi, a_lo);

    float s1[8], s2[8];
#pragma unroll
    for (int r = 0; r < 8; ++r) { s1[r] = 0.0f; s2[r] = 0.0f; }

    for (int nt = walker; nt < N_TILES; nt += n_walkers) {
        const int n0 = nt * 16;
        const float* brow = bank + (size_t)(n0 + lrow) * D_DIM;

        v8f acc = {};
#pragma unroll
        for (int ks = 0; ks < 4; ++ks) {
            const int kb = ks * 32 + hgrp * 16;
            v16h b_hi, b_lo;
#pragma unroll
            for (int j = 0; j < 16; ++j) {
                float x = brow[kb + j];
                _Float16 h = (_Float16)x;
                b_hi[j] = h;
                b_lo[j] = (_Float16)(x - (float)h);
            }
            acc = __builtin_amdgcn_wmma_f32_16x16x32_f16(
                false, a_hi[ks], false, b_hi, (short)0, acc, false, false);
            acc = __builtin_amdgcn_wmma_f32_16x16x32_f16(
                false, a_hi[ks], false, b_lo, (short)0, acc, false, false);
            acc = __builtin_amdgcn_wmma_f32_16x16x32_f16(
                false, a_lo[ks], false, b_hi, (short)0, acc, false, false);
        }

        const int n = n0 + lrow;
#pragma unroll
        for (int r = 0; r < 8; ++r) {
            const int mm = m_base + r + hgrp * 8;
            const size_t idx = (size_t)mm * N_COLS + n;
            const float e = __expf(acc[r] * INV_T);
            s1[r] += e * (float)bgm[idx];
            s2[r] += e * (float)inm[idx];
        }
    }

    laggr_reduce_commit(d1, d2, s1, s2, m_base, lrow, hgrp);
}

// ---------------------------------------------------------------------------
// Kernel 4: out = mean(log d1 - log d2)
// ---------------------------------------------------------------------------
__global__ void laggr_finish_kernel(const float* __restrict__ ws,
                                    float* __restrict__ out) {
    __shared__ float red[16];
    const int i = threadIdx.x;  // 512 threads = 16 waves
    float val = logf(ws[i]) - logf(ws[B_ROWS + i]);
#pragma unroll
    for (int off = 16; off > 0; off >>= 1) val += __shfl_xor(val, off, 32);
    if ((i & 31) == 0) red[i >> 5] = val;
    __syncthreads();
    if (i < 16) {
        float t = red[i];
#pragma unroll
        for (int off = 8; off > 0; off >>= 1) t += __shfl_xor(t, off, 16);
        if (i == 0) out[0] = t / (float)B_ROWS;
    }
}

extern "C" void kernel_launch(void* const* d_in, const int* in_sizes, int n_in,
                              void* d_out, int out_size, void* d_ws, size_t ws_size,
                              hipStream_t stream) {
    (void)in_sizes; (void)n_in; (void)out_size;
    const float* codes       = (const float*)d_in[0];
    const float* bank        = (const float*)d_in[1];
    const unsigned char* bgm = (const unsigned char*)d_in[2];   // bool mask
    const unsigned char* inm = (const unsigned char*)d_in[3];   // bool mask
    float* ws  = (float*)d_ws;
    float* out = (float*)d_out;

    laggr_prep_kernel<<<4, 256, 0, stream>>>(codes, ws);

    if (ws_size >= WS_FAST_BYTES) {
        _Float16* bankHi = (_Float16*)((char*)d_ws + WS_PLANE_OFF);
        _Float16* bankLo = bankHi + (size_t)N_COLS * D_DIM;
        laggr_split_kernel<<<4096, 256, 0, stream>>>(bank, bankHi, bankLo);
        // 2048 WGs x 4 waves = 8192 waves = 32 m-tiles x 256 n-walkers
        laggr_gemm_fast_kernel<<<2048, 128, 0, stream>>>(codes, bankHi, bankLo,
                                                         bgm, inm, ws);
    } else {
        laggr_gemm_fallback_kernel<<<2048, 128, 0, stream>>>(codes, bank,
                                                             bgm, inm, ws);
    }

    laggr_finish_kernel<<<1, 512, 0, stream>>>(ws, out);
}
